// PairwiseGrounder_77455440216103
// MI455X (gfx1250) — matrix operand order repaired and checked by reading, not compiled, not run on hardware
//
#include <hip/hip_runtime.h>
#include <hip/hip_bf16.h>

// Problem dims (fixed by the reference)
#define BB 64
#define NN 128
#define LL 64
#define DD 768
#define NEGV (-9e9f)

typedef __bf16 v16bf __attribute__((ext_vector_type(16)));
typedef __bf16 v8bf  __attribute__((ext_vector_type(8)));
typedef float  v8f   __attribute__((ext_vector_type(8)));

typedef __hip_bfloat16 bf16s;

// ---------------------------------------------------------------------------
// K0a: f32 -> bf16 (text embeddings)
// ---------------------------------------------------------------------------
__global__ __launch_bounds__(256) void k_cvt_text(const float* __restrict__ x,
                                                  bf16s* __restrict__ y, int n) {
  int k = blockIdx.x * 256 + threadIdx.x;
  if (k < n) y[k] = __float2bfloat16(x[k]);
}

// ---------------------------------------------------------------------------
// K0b: f32 -> bf16 image, both [B][L][D] and transposed [B][D][L]
// ---------------------------------------------------------------------------
__global__ __launch_bounds__(256) void k_cvt_img(const float* __restrict__ x,
                                                 bf16s* __restrict__ y,
                                                 bf16s* __restrict__ yT, int n) {
  int k = blockIdx.x * 256 + threadIdx.x;
  if (k >= n) return;
  int b = k / (LL * DD);
  int r = k - b * (LL * DD);
  int l = r / DD;
  int d = r - l * DD;
  bf16s v = __float2bfloat16(x[k]);
  y[k] = v;
  yT[((size_t)b * DD + d) * LL + l] = v;
}

// ---------------------------------------------------------------------------
// K1: per-(i,j) fused GEMM + masked double softmax -> S[i,j]; diagonal emits
//     row-softmax attention weights (bf16) for the i2s GEMM.
//     Block = 256 threads = 8 waves; wave w owns rows 16w..16w+15.
//     B (image[j]) is staged through LDS in double-buffered 64-wide K slabs
//     using CDNA5 async global->LDS copies (ASYNCcnt-tracked).
// ---------------------------------------------------------------------------
#define KSLAB 64
#define NSLAB (DD / KSLAB)   // 12
#define BPITCH 72            // bf16 elements; 36 dwords -> conflict-free columns

__global__ __launch_bounds__(256) void k_pair(const bf16s* __restrict__ textb,
                                              const bf16s* __restrict__ imgb,
                                              const int* __restrict__ tmask,
                                              const int* __restrict__ imask,
                                              float* __restrict__ S,
                                              bf16s* __restrict__ awout) {
  __shared__ __align__(16) __bf16 sB[2][LL][BPITCH];  // 2 x 64 x 144B = 18KB
  __shared__ float att[NN][LL + 1];                   // 33KB, padded rows
  __shared__ float red[2];
  __shared__ int s_tm[NN];
  __shared__ int s_im[LL];

  const int j = blockIdx.x, i = blockIdx.y;
  const int tid = threadIdx.x;
  const int lane = tid & 31;
  const int wave = tid >> 5;
  const int lo = lane & 15;   // row/col within 16
  const int hi = lane >> 4;   // half-wave selector

  if (tid < NN) s_tm[tid] = tmask[i * NN + tid];
  if (tid < LL) s_im[tid] = imask[j * LL + tid];
  if (tid == 0) { red[0] = 0.0f; red[1] = 0.0f; }

  const __bf16* gB = (const __bf16*)imgb + (size_t)j * LL * DD;

  // Stage slab s (image[j][0:64][s*64 : s*64+64], 8KB) into sB[buf].
  // 256 threads x 2 x async b128: fully coalesced, no VGPR round-trip.
  auto stage = [&](int s, int buf) {
#pragma unroll
    for (int c = 0; c < 2; ++c) {
      int q = tid * 2 + c;        // 16B chunk id, 0..511
      int row = q >> 3;           // image row l (0..63)
      int ce = (q & 7) * 8;       // bf16 element offset within slab row
      const __bf16* gp = gB + (size_t)row * DD + s * KSLAB + ce;
      unsigned lp = (unsigned)(uintptr_t)&sB[buf][row][ce];
      asm volatile("global_load_async_to_lds_b128 %0, %1, off"
                   :: "v"(lp), "v"(gp)
                   : "memory");
    }
  };

  // ---- GEMM: att_tile = text[i] (128x768) x image[j]^T (768x64), bf16 WMMA
  v8f c[4] = {};
  const __bf16* Arow =
      (const __bf16*)textb + ((size_t)(i * NN + wave * 16 + lo)) * DD;

  stage(0, 0);
  for (int s = 0; s < NSLAB; ++s) {
    if (s < NSLAB - 1) {
      stage(s + 1, (s + 1) & 1);                       // prefetch next slab
      asm volatile("s_wait_asynccnt 0x2" ::: "memory"); // slab s landed
    } else {
      asm volatile("s_wait_asynccnt 0x0" ::: "memory");
    }
    __syncthreads();  // every wave's async copies for slab s are visible
    const int buf = s & 1;
#pragma unroll
    for (int kk = 0; kk < 2; ++kk) {
      int k0 = s * KSLAB + kk * 32;
      // A frag (16x32 bf16): lane<16 holds K {0..15}, lane>=16 {16..31}
      union { v16bf v; v8bf h[2]; } ua;
      ua.v = *(const v16bf*)(Arow + k0 + hi * 16);
      v16bf a = ua.v;
#pragma unroll
      for (int t = 0; t < 4; ++t) {
        // B frag (32x16): column l = t*16+lo, 16 contiguous K per lane, from LDS
        const __bf16* bp = &sB[buf][t * 16 + lo][kk * 32 + hi * 16];
        v16bf b = *(const v16bf*)bp;
        c[t] = __builtin_amdgcn_wmma_f32_16x16x32_bf16(
            false, a, false, b, (short)0, c[t], false, false);
      }
    }
    __syncthreads();  // compute done before buf is re-staged (slab s+2)
  }

  // ---- apply reference masking: att = (mask && dot!=0) ? dot : NEG
#pragma unroll
  for (int t = 0; t < 4; ++t) {
    int col = t * 16 + lo;
    int mcol = s_im[col];
#pragma unroll
    for (int r = 0; r < 8; ++r) {
      int row = wave * 16 + hi * 8 + r;
      float v = c[t][r];
      int m = s_tm[row] * mcol;
      att[row][col] = (m != 0 && v != 0.0f) ? v : NEGV;
    }
  }
  __syncthreads();

  // ---- score1: row softmax over l (and diagonal attention weights)
  if (tid < NN) {
    int row = tid;
    int mrow = s_tm[row];
    float mx = -3.4e38f;
#pragma unroll 4
    for (int l = 0; l < LL; ++l) mx = fmaxf(mx, att[row][l]);
    float den = 0.0f, acc = 0.0f;
#pragma unroll 4
    for (int l = 0; l < LL; ++l) {
      float a = att[row][l];
      float e = __expf(a - mx);
      den += e;
      float m = (float)(mrow * s_im[l]);
      acc += e * m * a;
    }
    if (i == j) {
      float inv = 1.0f / den;
      bf16s* awr = awout + ((size_t)(i * NN + row)) * LL;
#pragma unroll 4
      for (int l = 0; l < LL; ++l)
        awr[l] = __float2bfloat16(__expf(att[row][l] - mx) * inv);
    }
    atomicAdd(&red[0], acc / den);
  }

  // ---- score2: column softmax over n
  if (tid < LL) {
    int col = tid;
    int mcol = s_im[col];
    float mx = -3.4e38f;
#pragma unroll 4
    for (int r = 0; r < NN; ++r) mx = fmaxf(mx, att[r][col]);
    float den = 0.0f, acc = 0.0f;
#pragma unroll 4
    for (int r = 0; r < NN; ++r) {
      float a = att[r][col];
      float e = __expf(a - mx);
      den += e;
      float m = (float)(s_tm[r] * mcol);
      acc += e * m * a;
    }
    atomicAdd(&red[1], acc / den);
  }

  __syncthreads();
  if (tid == 0) S[i * BB + j] = (red[0] + red[1]) * (1.0f / (float)NN);
}

// ---------------------------------------------------------------------------
// K2: text_emb_i2s[b] = aw[b] (128x64) @ image[b] (64x768) via bf16 WMMA.
//     B fragments read from the transposed image copy (contiguous K).
// ---------------------------------------------------------------------------
__global__ __launch_bounds__(256) void k_i2s(const bf16s* __restrict__ aw,
                                             const bf16s* __restrict__ imgT,
                                             float* __restrict__ out) {
  const int b = blockIdx.x;
  const int tid = threadIdx.x;
  const int lane = tid & 31;
  const int wave = tid >> 5;
  const int lo = lane & 15;
  const int hi = lane >> 4;

  // A frags (K = 64 -> two k-steps), hoisted: reused across all 48 n-tiles
  const __bf16* Ar = (const __bf16*)aw + ((size_t)(b * NN + wave * 16 + lo)) * LL;
  union { v16bf v; v8bf h[2]; } u0, u1;
  u0.v = *(const v16bf*)(Ar + hi * 16);
  u1.v = *(const v16bf*)(Ar + 32 + hi * 16);
  v16bf a0 = u0.v, a1 = u1.v;

  for (int nt = 0; nt < DD / 16; ++nt) {
    const __bf16* Bp = (const __bf16*)imgT + ((size_t)(b * DD + nt * 16 + lo)) * LL;
    v16bf b0 = *(const v16bf*)(Bp + hi * 16);
    v16bf b1 = *(const v16bf*)(Bp + 32 + hi * 16);
    v8f c = {};
    c = __builtin_amdgcn_wmma_f32_16x16x32_bf16(false, a0, false, b0, (short)0,
                                                c, false, false);
    c = __builtin_amdgcn_wmma_f32_16x16x32_bf16(false, a1, false, b1, (short)0,
                                                c, false, false);
#pragma unroll
    for (int r = 0; r < 8; ++r) {
      int row = wave * 16 + hi * 8 + r;
      out[((size_t)(b * NN + row)) * DD + nt * 16 + lo] = c[r];
    }
  }
}

// ---------------------------------------------------------------------------
// K3: loss = -(sum diag logsoftmax(S,rows) + sum diag logsoftmax(S,cols)) / B
// ---------------------------------------------------------------------------
__global__ __launch_bounds__(64) void k_loss(const float* __restrict__ S,
                                             float* __restrict__ out) {
  __shared__ float red;
  int t = threadIdx.x;
  if (t == 0) red = 0.0f;
  __syncthreads();

  float mr = -3.4e38f, mc = -3.4e38f;
  for (int k = 0; k < BB; ++k) {
    mr = fmaxf(mr, S[t * BB + k]);
    mc = fmaxf(mc, S[k * BB + t]);
  }
  float sr = 0.0f, sc = 0.0f;
  for (int k = 0; k < BB; ++k) {
    sr += __expf(S[t * BB + k] - mr);
    sc += __expf(S[k * BB + t] - mc);
  }
  float d = S[t * BB + t];
  float ls = (d - mr - __logf(sr)) + (d - mc - __logf(sc));
  atomicAdd(&red, ls);
  __syncthreads();
  if (t == 0) out[0] = -red / (float)BB;
}

// ---------------------------------------------------------------------------
extern "C" void kernel_launch(void* const* d_in, const int* in_sizes, int n_in,
                              void* d_out, int out_size, void* d_ws,
                              size_t ws_size, hipStream_t stream) {
  const float* text = (const float*)d_in[0];   // [64,128,768] f32
  const float* img  = (const float*)d_in[1];   // [64,64,768]  f32
  const int* tmask  = (const int*)d_in[2];     // [64,128]
  const int* imask  = (const int*)d_in[3];     // [64,64]

  float* out = (float*)d_out;                  // [0]=loss, [1..]=i2s [64,128,768]

  // Workspace layout (regions are naturally 256B-aligned)
  char* ws = (char*)d_ws;
  size_t off = 0;
  bf16s* textb = (bf16s*)(ws + off); off += (size_t)BB * NN * DD * 2;  // 12.6MB
  bf16s* imgb  = (bf16s*)(ws + off); off += (size_t)BB * LL * DD * 2;  //  6.3MB
  bf16s* imgT  = (bf16s*)(ws + off); off += (size_t)BB * DD * LL * 2;  //  6.3MB
  bf16s* aw    = (bf16s*)(ws + off); off += (size_t)BB * NN * LL * 2;  //  1.0MB
  float* S     = (float*)(ws + off); off += (size_t)BB * BB * 4;       //  16KB
  (void)ws_size; (void)n_in; (void)in_sizes; (void)out_size;

  const int nText = BB * NN * DD;  // 6,291,456
  const int nImg  = BB * LL * DD;  // 3,145,728

  k_cvt_text<<<(nText + 255) / 256, 256, 0, stream>>>(text, textb, nText);
  k_cvt_img<<<(nImg + 255) / 256, 256, 0, stream>>>(img, imgb, imgT, nImg);

  k_pair<<<dim3(BB, BB), 256, 0, stream>>>(textb, imgb, tmask, imask, S, aw);

  k_i2s<<<BB, 256, 0, stream>>>(aw, imgT, out + 1);

  k_loss<<<1, 64, 0, stream>>>(S, out);
}
